// PhysicalPooling_9981503996045
// MI455X (gfx1250) — matrix operand order repaired and checked by reading, not compiled, not run on hardware
//
#include <hip/hip_runtime.h>
#include <hip/hip_bf16.h>

typedef __attribute__((ext_vector_type(16))) _Float16 v16h;
typedef __attribute__((ext_vector_type(8)))  _Float16 v8h;
typedef __attribute__((ext_vector_type(2)))  _Float16 v2h;
typedef __attribute__((ext_vector_type(8)))  float    v8f;

#define PNUM 2048
#define BNUM 10000
#define NB   15
#define HDIM 64
#define MID  512
#define BOT  1024
#define NROWS (PNUM * NB)          // 30720
#define MTILES (NROWS / 64)        // 480

#define PI_F      3.14159265358979323846f
#define HALF_PI_F 1.57079632679489661923f

// ---------------- workspace layout (bytes, 1KB-aligned blocks) -------------
#define WS_ALIGN(x) (((x) + 1023) & ~(size_t)1023)
#define OFF_REL   ((size_t)0)                                   // NROWS*2 f32
#define OFF_H1    WS_ALIGN(OFF_REL  + (size_t)NROWS*2*4)        // P*MID f32 (4MB)
#define OFF_WEFF  WS_ALIGN(OFF_H1   + (size_t)PNUM*MID*4)       // 2*512 f32
#define OFF_BEFF  WS_ALIGN(OFF_WEFF + (size_t)2*MID*4)
#define OFF_BN1A  WS_ALIGN(OFF_BEFF + (size_t)MID*4)
#define OFF_BN1B  WS_ALIGN(OFF_BN1A + (size_t)MID*4)
#define OFF_BN2A  WS_ALIGN(OFF_BN1B + (size_t)MID*4)
#define OFF_BN2B  WS_ALIGN(OFF_BN2A + (size_t)BOT*4)
#define OFF_W2P   WS_ALIGN(OFF_BN2B + (size_t)BOT*4)            // 512*1024 f16 packed
#define OFF_PS    WS_ALIGN(OFF_W2P  + (size_t)MID*BOT*2)        // 480*1024 f32
#define OFF_PSS   WS_ALIGN(OFF_PS   + (size_t)MTILES*BOT*4)     // 480*1024 f32

// ---------------- k_prep: W_eff = W_sp @ W1_top, b_eff = b_sp@W1_top + b1 ---
__global__ __launch_bounds__(512) void k_prep(const float* __restrict__ W_sp,
                                              const float* __restrict__ b_sp,
                                              const float* __restrict__ W1,
                                              const float* __restrict__ b1,
                                              float* __restrict__ weff,
                                              float* __restrict__ beff) {
    int j = threadIdx.x;            // 0..511
    float we0 = 0.f, we1 = 0.f, bb = 0.f;
    #pragma unroll 8
    for (int t = 0; t < 64; ++t) {
        float w1v = W1[t * MID + j];
        we0 = fmaf(W_sp[t],      w1v, we0);
        we1 = fmaf(W_sp[64 + t], w1v, we1);
        bb  = fmaf(b_sp[t],      w1v, bb);
    }
    weff[j]       = we0;
    weff[MID + j] = we1;
    beff[j]       = bb + b1[j];
}

// ---------------- k_w2pack: W2 f32 -> f16, B-fragment order -----------------
// W2p[((kt*64 + nt)*32 + lane)*16 + e] = W2[(kt*32 + e + 16*(lane>>4))*BOT
//                                           + nt*16 + (lane&15)]
// so a lane's full 16x16x32 B-fragment is one contiguous, 32B-aligned v16h.
__global__ __launch_bounds__(256) void k_w2pack(const float* __restrict__ W2,
                                                _Float16* __restrict__ W2p) {
    int idx = blockIdx.x * 256 + threadIdx.x;   // exactly MID*BOT threads
    int e    = idx & 15;
    int lane = (idx >> 4) & 31;
    int nt   = (idx >> 9) & 63;
    int kt   = idx >> 15;
    int K = kt * 32 + e + 16 * (lane >> 4);
    int N = nt * 16 + (lane & 15);
    W2p[idx] = (_Float16)W2[(size_t)K * BOT + N];
}

// ---------------- k_polar: per-(ped,bin) min-radius boundary point ----------
// Boundary set (80KB) is streamed into LDS in 16KB chunks with
// global_load_async_to_lds_b128 (ASYNCcnt), then the transcendental-heavy
// pairwise loop runs out of LDS. key = (r_bits<<32)|idx; u64-min is
// lexicographic (min r, then min index) == pandas idxmin tie-break.
#define PCHUNK 2000                      // points per chunk (16000 B)
__global__ __launch_bounds__(256) void k_polar(const float* __restrict__ end_pos,
                                               const float* __restrict__ rel_pos,
                                               const float* __restrict__ bpts,
                                               float* __restrict__ rel) {
    __shared__ unsigned long long sbin[NB];
    __shared__ float sbp[PCHUNK * 2];    // staged boundary chunk
    const int p = blockIdx.x;
    const float ex = end_pos[2 * p], ey = end_pos[2 * p + 1];
    const float rx = rel_pos[2 * p], ry = rel_pos[2 * p + 1];
    const float thp = atan2f(ry, rx);
    const float cs = cosf(thp), sn = sinf(thp);

    const unsigned long long FB =
        ((unsigned long long)__float_as_uint(2.0f) << 32) | 0xFFFFFFFFull;
    if (threadIdx.x < NB) sbin[threadIdx.x] = FB;

    const float binscale = (float)NB / PI_F;   // 1/width
    const unsigned lds0 = (unsigned)(uintptr_t)&sbp[0];  // LDS byte offset
    const unsigned long long gbase = (unsigned long long)(uintptr_t)bpts;

    for (int c = 0; c < BNUM / PCHUNK; ++c) {
        const int base = c * PCHUNK;
        // async DMA chunk into LDS: 1000 x b128 (16B) transfers
        for (int j = threadIdx.x; j < (PCHUNK * 8) / 16; j += 256) {
            unsigned long long g = gbase + (size_t)base * 8 + (size_t)j * 16;
            unsigned l = lds0 + (unsigned)j * 16;
            asm volatile("global_load_async_to_lds_b128 %0, %1, off"
                         :: "v"(l), "v"(g) : "memory");
        }
        asm volatile("s_wait_asynccnt 0" ::: "memory");
        __syncthreads();

        for (int t = threadIdx.x; t < PCHUNK; t += 256) {
            float dx = sbp[2 * t] - ex, dy = sbp[2 * t + 1] - ey;
            float xb =  dx * cs + dy * sn;
            float yb = -dx * sn + dy * cs;
            float r  = sqrtf(xb * xb + yb * yb);
            float th = atan2f(yb, xb);
            if (th > -HALF_PI_F && th <= HALF_PI_F) {
                int bin = (int)floorf((th + HALF_PI_F) * binscale);
                bin = bin < 0 ? 0 : (bin > NB - 1 ? NB - 1 : bin);
                unsigned long long key =
                    ((unsigned long long)__float_as_uint(r) << 32) |
                    (unsigned)(base + t);
                atomicMin(&sbin[bin], key);  // ds_min_u64, order-independent
            }
        }
        __syncthreads();
    }

    if (threadIdx.x < NB) {
        int k = threadIdx.x;
        unsigned idx = (unsigned)sbin[k];
        float ox, oy;
        if (idx < (unsigned)BNUM) {          // winner rotates back to itself
            ox = bpts[2 * idx] - ex;
            oy = bpts[2 * idx + 1] - ey;
        } else {                             // fallback at bin center, r=NEIGH
            float ang = thp + (-HALF_PI_F + (PI_F / NB) * (k + 0.5f));
            ox = 2.0f * cosf(ang);
            oy = 2.0f * sinf(ang);
        }
        rel[(p * NB + k) * 2]     = ox;
        rel[(p * NB + k) * 2 + 1] = oy;
    }
}

// ---------------- k_h1: H1 = h @ W1_bot  (2048 x 64 x 512, WMMA f16) --------
__global__ __launch_bounds__(256) void k_h1(const float* __restrict__ h,
                                            const float* __restrict__ W1,
                                            float* __restrict__ H1) {
    const int wave = threadIdx.x >> 5, lane = threadIdx.x & 31;
    const int lm = lane & 15, lh = lane >> 4;
    const int gw = blockIdx.x * 8 + wave;    // 0..4095 wave-tiles
    const int tm = gw >> 5;                  // 0..127 (M tiles of 16)
    const int tn = gw & 31;                  // 0..31  (N tiles of 16)
    const int row  = tm * 16 + lm;
    const int coln = tn * 16 + lm;

    v8f acc = {};
    #pragma unroll
    for (int s = 0; s < 64; s += 32) {
        v16h af, bf;
        #pragma unroll
        for (int e = 0; e < 8; ++e)
            af[e] = (_Float16)h[row * HDIM + s + e + 16 * lh];
        #pragma unroll
        for (int e = 8; e < 16; ++e)
            af[e] = (_Float16)h[row * HDIM + s + e + 16 * lh];
        #pragma unroll
        for (int e = 0; e < 16; ++e)
            bf[e] = (_Float16)W1[(64 + s + e + 16 * lh) * MID + coln];
        acc = __builtin_amdgcn_wmma_f32_16x16x32_f16(
            false, af, false, bf, (short)0, acc, false, false);
    }
    #pragma unroll
    for (int r = 0; r < 8; ++r)
        H1[(tm * 16 + r + 8 * lh) * MID + coln] = acc[r];
}

// ---------------- k_y1_stats: BN1 column stats without materializing Y1 -----
__global__ __launch_bounds__(256) void k_y1_stats(const float* __restrict__ rel,
                                                  const float* __restrict__ H1,
                                                  const float* __restrict__ weff,
                                                  const float* __restrict__ beff,
                                                  const float* __restrict__ g1,
                                                  const float* __restrict__ be1,
                                                  float* __restrict__ bn1a,
                                                  float* __restrict__ bn1b) {
    const int j = blockIdx.x * 256 + threadIdx.x;   // 0..511 (one column each)
    const float we0 = weff[j], we1 = weff[MID + j], be = beff[j];
    float s = 0.f, ss = 0.f;
    for (int p = 0; p < PNUM; ++p) {
        float base = H1[(size_t)p * MID + j] + be;
        #pragma unroll
        for (int k = 0; k < NB; ++k) {
            int i = p * NB + k;
            float y = fmaf(rel[2 * i], we0, fmaf(rel[2 * i + 1], we1, base));
            s += y;
            ss = fmaf(y, y, ss);
        }
    }
    const float inv = 1.0f / (float)NROWS;
    float m = s * inv;
    float v = fmaf(-m, m, ss * inv);
    float a = g1[j] * rsqrtf(v + 1e-5f);
    bn1a[j] = a;
    bn1b[j] = be1[j] - a * m;                // A1 = relu(a*y + b)
}

// ---------------- k_gemm2: 30720 x 512 x 1024, WMMA f16, two passes ---------
// A1 tile built on the fly in LDS (paired f16 stores); A-fragments are two
// aligned ds_load_b128; B-fragments are direct v16h global loads from the
// pre-packed W2p (L2-resident). pass 1: deterministic per-block-row column
// sum/sum-sq of Y2. pass 2: BN2+ReLU+max-over-15-bins via u32 atomicMax.
__global__ __launch_bounds__(256) void k_gemm2(const float* __restrict__ rel,
                                               const float* __restrict__ H1,
                                               const float* __restrict__ weff,
                                               const float* __restrict__ beff,
                                               const float* __restrict__ bn1a,
                                               const float* __restrict__ bn1b,
                                               const _Float16* __restrict__ W2p,
                                               const float* __restrict__ b2,
                                               float* __restrict__ psum,
                                               float* __restrict__ psumsq,
                                               const float* __restrict__ bn2a,
                                               const float* __restrict__ bn2b,
                                               unsigned* __restrict__ outp,
                                               int pass) {
    __shared__ _Float16 lA[64][32];      // A1 tile (f16), rows 64B apart
    __shared__ float sred[2][256];
    __shared__ float ssred[2][256];

    const int tid = threadIdx.x;
    const int wave = tid >> 5, lane = tid & 31;
    const int lm = lane & 15, lh = lane >> 4;
    const int gm0 = blockIdx.y * 64;     // row base (of 30720)
    const int gn0 = blockIdx.x * 64;     // col base (of 1024)

    v8f acc[2] = {};

    for (int ks = 0; ks < MID; ks += 32) {
        const int kt = ks >> 5;
        if (ks + 32 < MID)               // global_prefetch_b8 next W2p slice
            __builtin_prefetch(
                &W2p[(((size_t)(kt + 1) * 64) + blockIdx.x * 4) * 32 * 16], 0, 1);
        // stage A1 (64 x 32) as 1024 paired-f16 (u32) stores
        #pragma unroll
        for (int u = 0; u < 4; ++u) {
            int idx = u * 256 + tid;         // u32 index
            int r  = idx >> 4;               // 16 u32 per row
            int kk = (idx & 15) * 2;
            int i = gm0 + r;
            int p = i / NB;
            float rx = rel[2 * i], ry = rel[2 * i + 1];
            int j = ks + kk;
            float y0 = fmaf(rx, weff[j],
                       fmaf(ry, weff[MID + j],
                            H1[(size_t)p * MID + j] + beff[j]));
            float y1 = fmaf(rx, weff[j + 1],
                       fmaf(ry, weff[MID + j + 1],
                            H1[(size_t)p * MID + j + 1] + beff[j + 1]));
            v2h pk;
            pk[0] = (_Float16)fmaxf(fmaf(bn1a[j],     y0, bn1b[j]),     0.f);
            pk[1] = (_Float16)fmaxf(fmaf(bn1a[j + 1], y1, bn1b[j + 1]), 0.f);
            *(v2h*)&lA[r][kk] = pk;
        }
        __syncthreads();
        #pragma unroll
        for (int q = 0; q < 2; ++q) {
            int t = wave * 2 + q;
            int msub = t >> 2, nsub = t & 3;
            int rowl = msub * 16 + lm;
            // A fragment: two aligned ds_load_b128
            v8h alo = *(const v8h*)&lA[rowl][16 * lh];
            v8h ahi = *(const v8h*)&lA[rowl][16 * lh + 8];
            v16h af = __builtin_shufflevector(alo, ahi,
                0, 1, 2, 3, 4, 5, 6, 7, 8, 9, 10, 11, 12, 13, 14, 15);
            // B fragment: one contiguous 32B global load from packed W2
            int ntg = blockIdx.x * 4 + nsub;
            v16h bf = *(const v16h*)&W2p[(((size_t)kt * 64 + ntg) * 32 + lane) * 16];
            acc[q] = __builtin_amdgcn_wmma_f32_16x16x32_f16(
                false, af, false, bf, (short)0, acc[q], false, false);
        }
        __syncthreads();
    }

    // epilogue
    #pragma unroll
    for (int q = 0; q < 2; ++q) {
        int t = wave * 2 + q;
        int msub = t >> 2, nsub = t & 3;
        int colg = gn0 + nsub * 16 + lm;
        float b2c = b2[colg];
        if (pass == 1) {
            float s = 0.f, ssq = 0.f;
            #pragma unroll
            for (int r = 0; r < 8; ++r) {
                float y2 = acc[q][r] + b2c;
                s += y2;
                ssq = fmaf(y2, y2, ssq);
            }
            sred[q][tid]  = s;
            ssred[q][tid] = ssq;
        } else {
            float a2 = bn2a[colg], bb = bn2b[colg];
            #pragma unroll
            for (int r = 0; r < 8; ++r) {
                float y2 = acc[q][r] + b2c;
                float z  = fmaxf(fmaf(a2, y2, bb), 0.f);
                int rowg = gm0 + msub * 16 + r + 8 * lh;
                atomicMax(&outp[(size_t)(rowg / NB) * BOT + colg],
                          __float_as_uint(z));   // z >= 0 -> bits monotone
            }
        }
    }
    if (pass == 1) {
        __syncthreads();
        if (tid < 64) {            // fixed-order gather -> deterministic sums
            int g = tid >> 4, cl = tid & 15;
            float s = 0.f, ssq = 0.f;
            #pragma unroll
            for (int jj = 0; jj < 4; ++jj) {
                int t = g + 4 * jj;
                int w = t >> 1, q = t & 1;
                #pragma unroll
                for (int lh2 = 0; lh2 < 2; ++lh2) {
                    int src = w * 32 + lh2 * 16 + cl;
                    s   += sred[q][src];
                    ssq += ssred[q][src];
                }
            }
            psum[(size_t)blockIdx.y * BOT + gn0 + tid]   = s;
            psumsq[(size_t)blockIdx.y * BOT + gn0 + tid] = ssq;
        }
    }
}

// ---------------- k_bn2: reduce 480 partials per column (fixed order) -------
__global__ __launch_bounds__(256) void k_bn2(const float* __restrict__ psum,
                                             const float* __restrict__ psumsq,
                                             const float* __restrict__ g2,
                                             const float* __restrict__ be2,
                                             float* __restrict__ bn2a,
                                             float* __restrict__ bn2b) {
    const int j = blockIdx.x * 256 + threadIdx.x;   // 0..1023
    float s = 0.f, ss = 0.f;
    for (int bm = 0; bm < MTILES; ++bm) {
        s  += psum[(size_t)bm * BOT + j];
        ss += psumsq[(size_t)bm * BOT + j];
    }
    const float inv = 1.0f / (float)NROWS;
    float m = s * inv;
    float v = fmaf(-m, m, ss * inv);
    float a = g2[j] * rsqrtf(v + 1e-5f);
    bn2a[j] = a;
    bn2b[j] = be2[j] - a * m;
}

// ---------------- k_zero: out = 0 (needed for atomicMax of relu values) -----
__global__ __launch_bounds__(256) void k_zero(unsigned* __restrict__ outp) {
    outp[(size_t)blockIdx.x * 256 + threadIdx.x] = 0u;
}

extern "C" void kernel_launch(void* const* d_in, const int* in_sizes, int n_in,
                              void* d_out, int out_size, void* d_ws, size_t ws_size,
                              hipStream_t stream) {
    const float* h_states = (const float*)d_in[0];
    const float* end_pos  = (const float*)d_in[1];
    const float* rel_pos  = (const float*)d_in[2];
    const float* bpts     = (const float*)d_in[3];
    const float* W_sp     = (const float*)d_in[4];
    const float* b_sp     = (const float*)d_in[5];
    const float* W1       = (const float*)d_in[6];
    const float* b1       = (const float*)d_in[7];
    const float* g1       = (const float*)d_in[8];
    const float* be1      = (const float*)d_in[9];
    const float* W2       = (const float*)d_in[10];
    const float* b2       = (const float*)d_in[11];
    const float* g2       = (const float*)d_in[12];
    const float* be2      = (const float*)d_in[13];

    char* ws = (char*)d_ws;
    float*    rel  = (float*)(ws + OFF_REL);
    float*    H1   = (float*)(ws + OFF_H1);
    float*    weff = (float*)(ws + OFF_WEFF);
    float*    beff = (float*)(ws + OFF_BEFF);
    float*    bn1a = (float*)(ws + OFF_BN1A);
    float*    bn1b = (float*)(ws + OFF_BN1B);
    float*    bn2a = (float*)(ws + OFF_BN2A);
    float*    bn2b = (float*)(ws + OFF_BN2B);
    _Float16* W2p  = (_Float16*)(ws + OFF_W2P);
    float*    ps   = (float*)(ws + OFF_PS);
    float*    pss  = (float*)(ws + OFF_PSS);

    k_prep  <<<1, 512, 0, stream>>>(W_sp, b_sp, W1, b1, weff, beff);
    k_w2pack<<<(MID * BOT) / 256, 256, 0, stream>>>(W2, W2p);
    k_polar <<<PNUM, 256, 0, stream>>>(end_pos, rel_pos, bpts, rel);
    k_h1    <<<512, 256, 0, stream>>>(h_states, W1, H1);
    k_y1_stats<<<2, 256, 0, stream>>>(rel, H1, weff, beff, g1, be1, bn1a, bn1b);
    k_gemm2 <<<dim3(BOT / 64, MTILES), 256, 0, stream>>>(
        rel, H1, weff, beff, bn1a, bn1b, W2p, b2,
        ps, pss, nullptr, nullptr, nullptr, 1);
    k_bn2   <<<BOT / 256, 256, 0, stream>>>(ps, pss, g2, be2, bn2a, bn2b);
    k_zero  <<<(PNUM * BOT) / 256, 256, 0, stream>>>((unsigned*)d_out);
    k_gemm2 <<<dim3(BOT / 64, MTILES), 256, 0, stream>>>(
        rel, H1, weff, beff, bn1a, bn1b, W2p, b2,
        nullptr, nullptr, bn2a, bn2b, (unsigned*)d_out, 2);
}
